// MalariaGCN_21251498181391
// MI455X (gfx1250) — matrix operand order repaired and checked by reading, not compiled, not run on hardware
//
#include <hip/hip_runtime.h>
#include <hip/hip_bf16.h>

typedef __attribute__((ext_vector_type(16))) _Float16 v16h;
typedef __attribute__((ext_vector_type(8)))  float    v8f;

#define DH 128

// ---------------------------------------------------------------- utilities
__global__ void zero_f32_kernel(float* __restrict__ p, long long n) {
    long long i = (long long)blockIdx.x * blockDim.x + threadIdx.x;
    if (i < n) p[i] = 0.0f;
}

__global__ void degree_kernel(const int* __restrict__ col, float* __restrict__ deg, int E) {
    int e = blockIdx.x * blockDim.x + threadIdx.x;
    if (e < E) atomicAdd(&deg[col[e]], 1.0f);
}

__global__ void dinv_kernel(float* __restrict__ deg, int n) {
    int i = blockIdx.x * blockDim.x + threadIdx.x;
    if (i < n) {
        float d = deg[i];
        deg[i] = (d > 0.0f) ? rsqrtf(d) : 0.0f;   // deg>=1 when >0, matches ref
    }
}

// ---------------------------------------------------------------- WMMA GEMM
// C[n_rows,128] = A[n_rows,128] @ W[128,128] (+ bias), f32 in/out,
// f16 WMMA with f32 accumulation (v_wmma_f32_16x16x32_f16).
// Block: 256 threads = 8 waves; wave w computes rows [blk*128 + w*16, +16) x all 128 cols.
__global__ __launch_bounds__(256) void gemm128_wmma_kernel(
    const float* __restrict__ A, const float* __restrict__ W,
    const float* __restrict__ bias, float* __restrict__ C, int n_rows)
{
    // B fragments pre-swizzled: [kb][nb][lane][half]  (4*8*32*16 f16 = 32 KB)
    __shared__ __align__(32) _Float16 Bs[4][8][32][16];

    const int tid = threadIdx.x;
    // Cooperative fill. B 32x16 f16 layout used here (contiguous k, same order as A):
    //   lane 0-15 : N = lane,    halves h -> K = h      (0..15)
    //   lane 16-31: N = lane-16, halves h -> K = 16 + h (16..31)
    for (int s = tid; s < 4 * 8 * 32; s += 256) {
        const int kb   = s >> 8;          // 0..3
        const int nb   = (s >> 5) & 7;    // 0..7
        const int lane = s & 31;
        const int n     = nb * 16 + (lane & 15);
        const int kbase = kb * 32 + ((lane < 16) ? 0 : 16);
#pragma unroll
        for (int h = 0; h < 16; ++h)
            Bs[kb][nb][lane][h] = (_Float16)W[(kbase + h) * DH + n];
    }
    __syncthreads();

    const int wave = tid >> 5;
    const int lane = tid & 31;
    const int m0 = (blockIdx.x * 8 + wave) * 16;
    if (m0 >= n_rows) return;            // wave-uniform: EXEC stays all-ones for WMMA

    v8f acc[8];
#pragma unroll
    for (int nb = 0; nb < 8; ++nb) acc[nb] = (v8f){0,0,0,0,0,0,0,0};

    // A 16x32 f16 layout: lane 0-15 row M=lane holds K {0..15};
    // lanes 16-31 same rows, K {16..31}.
    const float* arow = A + (long long)(m0 + (lane & 15)) * DH;
    const int khi = (lane < 16) ? 0 : 16;

#pragma unroll
    for (int kb = 0; kb < 4; ++kb) {
        v16h a;
#pragma unroll
        for (int h = 0; h < 16; ++h) {
            const int kl = khi + h;
            a[h] = (_Float16)arow[kb * 32 + kl];
        }
#pragma unroll
        for (int nb = 0; nb < 8; ++nb) {
            v16h b = *(const v16h*)(&Bs[kb][nb][lane][0]);
            acc[nb] = __builtin_amdgcn_wmma_f32_16x16x32_f16(
                /*neg_a=*/false, a, /*neg_b=*/false, b,
                /*c_mod=*/(short)0, acc[nb], /*reuse_a=*/false, /*reuse_b=*/false);
        }
    }

    // C/D layout: VGPR r -> row m0 + r (lanes 0-15) / m0 + 8 + r (lanes 16-31), N = lane&15.
    const int mrow = m0 + ((lane >= 16) ? 8 : 0);
#pragma unroll
    for (int nb = 0; nb < 8; ++nb) {
        const int ncol = nb * 16 + (lane & 15);
        const float bv = bias ? bias[ncol] : 0.0f;
#pragma unroll
        for (int r = 0; r < 8; ++r)
            C[(long long)(mrow + r) * DH + ncol] = acc[nb][r] + bv;
    }
}

// ---------------------------------------------------------------- edge scatter
// out[col[e],:] += H[row[e],:] * dinv[row]*dinv[col];  32 threads per edge, float4 each.
__global__ void scatter_kernel(const float* __restrict__ H,
                               const int* __restrict__ rowi, const int* __restrict__ coli,
                               const float* __restrict__ dinv, float* __restrict__ out, int E)
{
    long long idx = (long long)blockIdx.x * blockDim.x + threadIdx.x;
    if (idx >= (long long)E * 32) return;
    const int e  = (int)(idx >> 5);
    const int d4 = (int)(idx & 31) * 4;
    const int r = rowi[e], c = coli[e];
    const float w = dinv[r] * dinv[c];
    const float4 v = *(const float4*)(H + (long long)r * DH + d4);
    float* o = out + (long long)c * DH + d4;
    atomicAdd(o + 0, v.x * w);
    atomicAdd(o + 1, v.y * w);
    atomicAdd(o + 2, v.z * w);
    atomicAdd(o + 3, v.w * w);
}

// ---------------------------------------------------------------- LN + ReLU (+res)
// One wave per node (wave32); each lane owns 4 of the 128 features.
__global__ __launch_bounds__(256) void post_kernel(float* __restrict__ H,
    const float* __restrict__ bias, const float* __restrict__ g, const float* __restrict__ b,
    const float* __restrict__ res, int n)
{
    const int wave = threadIdx.x >> 5, lane = threadIdx.x & 31;
    const int node = blockIdx.x * 8 + wave;
    if (node >= n) return;
    const int d0 = lane * 4;
    float* hp = H + (long long)node * DH + d0;

    float v[4];
#pragma unroll
    for (int i = 0; i < 4; ++i) v[i] = hp[i] + bias[d0 + i];

    float s = v[0] + v[1] + v[2] + v[3];
#pragma unroll
    for (int off = 16; off > 0; off >>= 1) s += __shfl_xor(s, off, 32);
    const float mean = s * (1.0f / 128.0f);

    float sq = 0.0f;
#pragma unroll
    for (int i = 0; i < 4; ++i) { const float d = v[i] - mean; sq += d * d; }
#pragma unroll
    for (int off = 16; off > 0; off >>= 1) sq += __shfl_xor(sq, off, 32);
    const float inv = rsqrtf(sq * (1.0f / 128.0f) + 1e-5f);

#pragma unroll
    for (int i = 0; i < 4; ++i) {
        float o = (v[i] - mean) * inv * g[d0 + i] + b[d0 + i];
        o = fmaxf(o, 0.0f);                                   // ReLU
        if (res) o += res[(long long)node * DH + d0 + i];     // skip connection
        hp[i] = o;
    }
}

// ---------------------------------------------------------------- MLP head
// h(128) -> relu(fc1: 64) -> fc2: 1 -> softplus.  One wave per node, 2 hidden/lane.
__global__ __launch_bounds__(256) void head_kernel(const float* __restrict__ H,
    const float* __restrict__ w1, const float* __restrict__ b1,
    const float* __restrict__ w2, const float* __restrict__ b2,
    float* __restrict__ out, int n)
{
    const int wave = threadIdx.x >> 5, lane = threadIdx.x & 31;
    const int node = blockIdx.x * 8 + wave;
    if (node >= n) return;
    const float* h = H + (long long)node * DH;
    const int j0 = lane * 2, j1 = j0 + 1;

    float a0 = b1[j0], a1 = b1[j1];
    for (int k = 0; k < DH; ++k) {
        const float xv = h[k];                 // wave-uniform broadcast load
        a0 = fmaf(xv, w1[k * 64 + j0], a0);
        a1 = fmaf(xv, w1[k * 64 + j1], a1);
    }
    a0 = fmaxf(a0, 0.0f);
    a1 = fmaxf(a1, 0.0f);

    float p = a0 * w2[j0] + a1 * w2[j1];
#pragma unroll
    for (int off = 16; off > 0; off >>= 1) p += __shfl_xor(p, off, 32);

    if (lane == 0) {
        const float z = p + b2[0];
        out[node] = fmaxf(z, 0.0f) + log1pf(expf(-fabsf(z)));   // stable softplus
    }
}

// ---------------------------------------------------------------- launcher
extern "C" void kernel_launch(void* const* d_in, const int* in_sizes, int n_in,
                              void* d_out, int out_size, void* d_ws, size_t ws_size,
                              hipStream_t stream)
{
    const float* x       = (const float*)d_in[0];
    const int*   ei      = (const int*)  d_in[1];
    const float* conv1_w = (const float*)d_in[2];
    const float* conv1_b = (const float*)d_in[3];
    const float* ln1_g   = (const float*)d_in[4];
    const float* ln1_b   = (const float*)d_in[5];
    const float* conv2_w = (const float*)d_in[6];
    const float* conv2_b = (const float*)d_in[7];
    const float* ln2_g   = (const float*)d_in[8];
    const float* ln2_b   = (const float*)d_in[9];
    const float* skip_w  = (const float*)d_in[10];
    const float* skip_b  = (const float*)d_in[11];
    const float* fc1_w   = (const float*)d_in[12];
    const float* fc1_b   = (const float*)d_in[13];
    const float* fc2_w   = (const float*)d_in[14];
    const float* fc2_b   = (const float*)d_in[15];
    float*       y       = (float*)d_out;

    const int N = in_sizes[0] / DH;
    const int E = in_sizes[1] / 2;
    const int* row = ei;
    const int* col = ei + E;

    // Workspace layout (floats): dinv[N] | R | G | S | T  (each N*128)
    float* ws   = (float*)d_ws;
    const long long feat = (long long)N * DH;
    float* dinv = ws;
    float* R    = ws + (((long long)N + 63) & ~63LL);
    float* G    = R + feat;
    float* S    = G + feat;
    float* T    = S + feat;

    const int TB = 256;
    const int gN     = (N + TB - 1) / TB;
    const int gE     = (E + TB - 1) / TB;
    const int gFeat  = (int)((feat + TB - 1) / TB);
    const int gScat  = (int)(((long long)E * 32 + TB - 1) / TB);
    const int gNode  = (N + 7) / 8;                    // wave-per-node kernels
    const int gGemm  = ((N + 15) / 16 + 7) / 8;        // 8 wave-tiles per block

    // normalization coefficients
    zero_f32_kernel<<<gN, TB, 0, stream>>>(dinv, N);
    degree_kernel  <<<gE, TB, 0, stream>>>(col, dinv, E);
    dinv_kernel    <<<gN, TB, 0, stream>>>(dinv, N);

    // residual = x @ skip_w + skip_b
    gemm128_wmma_kernel<<<gGemm, TB, 0, stream>>>(x, skip_w, skip_b, R, N);

    // conv1: G = x @ conv1_w; S = scatter(G); S = relu(LN(S + b1))
    gemm128_wmma_kernel<<<gGemm, TB, 0, stream>>>(x, conv1_w, nullptr, G, N);
    zero_f32_kernel    <<<gFeat, TB, 0, stream>>>(S, feat);
    scatter_kernel     <<<gScat, TB, 0, stream>>>(G, row, col, dinv, S, E);
    post_kernel        <<<gNode, TB, 0, stream>>>(S, conv1_b, ln1_g, ln1_b, nullptr, N);

    // conv2: G = S @ conv2_w; T = scatter(G); T = relu(LN(T + b2)) + residual
    gemm128_wmma_kernel<<<gGemm, TB, 0, stream>>>(S, conv2_w, nullptr, G, N);
    zero_f32_kernel    <<<gFeat, TB, 0, stream>>>(T, feat);
    scatter_kernel     <<<gScat, TB, 0, stream>>>(G, row, col, dinv, T, E);
    post_kernel        <<<gNode, TB, 0, stream>>>(T, conv2_b, ln2_g, ln2_b, R, N);

    // MLP head + softplus
    head_kernel<<<gNode, TB, 0, stream>>>(T, fc1_w, fc1_b, fc2_w, fc2_b, y, N);
}
